// HybridPlanningModel_23673859736031
// MI455X (gfx1250) — matrix operand order repaired and checked by reading, not compiled, not run on hardware
//
#include <hip/hip_runtime.h>
#include <hip/hip_bf16.h>

// ---------------------------------------------------------------------------
// HybridPlanningModel for MI455X (gfx1250, wave32).
// Dense GEMMs: v_wmma_f32_16x16x32_bf16, one wave = 16 x (16*NT) output tile
// (NT=4 normally) so each A fragment feeds NT WMMAs -> ~13 FLOP/byte operand
// traffic, weights/A panels resident in 192MB L2.
// Irregular parts (edge softmax-scatter, GRU gating, 10x10 attention,
// group-norm reductions) are VALU kernels with f32 atomics.
// Scratch arena lives in d_ws (~184 MB), 256B-aligned offsets.
// ---------------------------------------------------------------------------

#define BB   4
#define WW   10
#define NN   1000
#define BW   40            // BB*WW
#define M40  40000         // BB*WW*NN
#define EE   16000
#define ET   17000         // EE + NN self loops
#define ETP  17008         // padded to /16
#define DD   128
#define HH   4
#define FHH  32
#define MGRU 4000          // BB*NN

typedef __attribute__((ext_vector_type(16))) __bf16          v16bf;
typedef __attribute__((ext_vector_type(8)))  float           v8f;
typedef __attribute__((ext_vector_type(8)))  unsigned short  u16x8;

union BfFrag { u16x8 u[2]; v16bf v; };

__device__ __forceinline__ unsigned short f2bf(float f) {
  unsigned u = __float_as_uint(f);
  u += 0x7FFFu + ((u >> 16) & 1u);
  return (unsigned short)(u >> 16);
}
__device__ __forceinline__ float sigm(float x) { return 1.f / (1.f + __expf(-x)); }

// ---------------------------------------------------------------------------
// WMMA GEMM: C[M x N] = A[M x Kp](bf16) * Bt[N x Kp](bf16)^T + bias, opt ReLU.
// One wave computes a 16 x (16*NT) tile; A fragment reused across NT WMMAs.
// Fragment layouts per ISA 7.12.2 (16-bit A / B, f32 C/D).
// ---------------------------------------------------------------------------
template <int NT>
__global__ void k_wmma_gemm(const unsigned short* __restrict__ A,
                            const unsigned short* __restrict__ Bt,
                            float* __restrict__ C, const float* __restrict__ bias,
                            int M, int N, int Kp, int relu)
{
  const int wave = threadIdx.x >> 5;
  const int lane = threadIdx.x & 31;
  const int grp  = blockIdx.x * (blockDim.x >> 5) + wave;
  const int Ngr  = N / (16 * NT);
  const int mt   = grp / Ngr;
  const int ng   = grp - mt * Ngr;
  const int Mtl  = (M + 15) >> 4;
  if (mt >= Mtl) return;                       // wave-uniform: EXEC stays all-ones

  const int hi  = lane >> 4;
  const int l15 = lane & 15;
  const unsigned short* arow = A + (size_t)(mt * 16 + l15) * Kp + hi * 8;
  const unsigned short* brow[NT];
#pragma unroll
  for (int i = 0; i < NT; ++i)
    brow[i] = Bt + (size_t)((ng * NT + i) * 16 + l15) * Kp + hi * 8;

  v8f c[NT];
#pragma unroll
  for (int i = 0; i < NT; ++i) c[i] = (v8f){};

  for (int kk = 0; kk < Kp; kk += 32) {
    __builtin_prefetch(arow + kk + 64, 0, 1);  // global_prefetch_b8, next K-slab
    BfFrag a;
    a.u[0] = *(const u16x8*)(arow + kk);       // K = kk + hi*8 + [0..7]
    a.u[1] = *(const u16x8*)(arow + kk + 16);  // K = kk + 16 + hi*8 + [0..7]
#pragma unroll
    for (int i = 0; i < NT; ++i) {
      BfFrag b;
      b.u[0] = *(const u16x8*)(brow[i] + kk);      // K = kk + hi*8 + [0..7]
      b.u[1] = *(const u16x8*)(brow[i] + kk + 16); // K = kk + 16 + hi*8 + [0..7]
      c[i] = __builtin_amdgcn_wmma_f32_16x16x32_bf16(false, a.v, false, b.v,
                                                     (short)0, c[i], false, false);
    }
  }
#pragma unroll
  for (int i = 0; i < NT; ++i) {
    const int col = (ng * NT + i) * 16 + l15;
    const float bv = bias ? bias[col] : 0.f;
#pragma unroll
    for (int j = 0; j < 8; ++j) {
      int row = mt * 16 + hi * 8 + j;          // f32 C/D layout: VGPRj -> M=j(+8)
      if (row < M) {
        float v = c[i][j] + bv;
        if (relu) v = fmaxf(v, 0.f);
        C[(size_t)row * N + col] = v;
      }
    }
  }
}

// ---------------------------------------------------------------------------
// bf16 operand converters (zero-padded)
// ---------------------------------------------------------------------------
__global__ void k_conv_a(const float* __restrict__ A, unsigned short* __restrict__ O,
                         int M, int Mp, int K, int Kp)
{
  long idx = blockIdx.x * (long)blockDim.x + threadIdx.x;
  if (idx >= (long)Mp * Kp) return;
  int k = idx % Kp; long m = idx / Kp;
  O[idx] = (m < M && k < K) ? f2bf(A[m * K + k]) : (unsigned short)0;
}
// W is (K x N): Bt[n][k] = W[k][n]
__global__ void k_conv_w_t(const float* __restrict__ Wm, unsigned short* __restrict__ O,
                           int K, int N, int Kp)
{
  int idx = blockIdx.x * blockDim.x + threadIdx.x;
  if (idx >= N * Kp) return;
  int k = idx % Kp, n = idx / Kp;
  O[idx] = (k < K) ? f2bf(Wm[k * N + n]) : (unsigned short)0;
}
// W is (N x K) already in Bt orientation (used for x @ W.T)
__global__ void k_conv_w(const float* __restrict__ Wm, unsigned short* __restrict__ O,
                         int N, int K, int Kp)
{
  int idx = blockIdx.x * blockDim.x + threadIdx.x;
  if (idx >= N * Kp) return;
  int k = idx % Kp, n = idx / Kp;
  O[idx] = (k < K) ? f2bf(Wm[n * K + k]) : (unsigned short)0;
}

// Build A0 = concat(x_temporal, struct_emb) as bf16, Kp=96 (72 padded)
__global__ void k_build_x(const float* __restrict__ xt, const float* __restrict__ se,
                          unsigned short* __restrict__ A)
{
  int idx = blockIdx.x * blockDim.x + threadIdx.x;
  if (idx >= M40 * 96) return;
  int col = idx % 96, row = idx / 96;
  float v = 0.f;
  if (col < 8)       v = xt[row * 8 + col];
  else if (col < 72) v = se[(row % NN) * 64 + (col - 8)];
  A[idx] = f2bf(v);
}

// ---------------------------------------------------------------------------
// Row LayerNorm (D=128), optional residual input, optional ReLU. Wave per row.
// ---------------------------------------------------------------------------
__global__ void k_ln(const float* __restrict__ x, const float* __restrict__ res,
                     float* __restrict__ out, const float* __restrict__ g,
                     const float* __restrict__ b, int M, int relu)
{
  int row = blockIdx.x * (blockDim.x >> 5) + (threadIdx.x >> 5);
  if (row >= M) return;
  int lane = threadIdx.x & 31;
  const float* xr = x + (size_t)row * DD;
  float v[4], s = 0.f;
#pragma unroll
  for (int i = 0; i < 4; ++i) {
    int d = lane + 32 * i;
    v[i] = xr[d] + (res ? res[(size_t)row * DD + d] : 0.f);
    s += v[i];
  }
  for (int o = 16; o > 0; o >>= 1) s += __shfl_xor(s, o, 32);
  float mu = s * (1.f / DD), q = 0.f;
#pragma unroll
  for (int i = 0; i < 4; ++i) { float d0 = v[i] - mu; q += d0 * d0; }
  for (int o = 16; o > 0; o >>= 1) q += __shfl_xor(q, o, 32);
  float inv = rsqrtf(q * (1.f / DD) + 1e-5f);
#pragma unroll
  for (int i = 0; i < 4; ++i) {
    int d = lane + 32 * i;
    float o2 = g[d] * (v[i] - mu) * inv + b[d];
    if (relu) o2 = fmaxf(o2, 0.f);
    out[(size_t)row * DD + d] = o2;
  }
}

// ---------------------------------------------------------------------------
// Edge embedding (K=4, VALU) + column mean + mean-row fill
// ---------------------------------------------------------------------------
__global__ void k_edge_embed(const float* __restrict__ ea, const float* __restrict__ w,
                             const float* __restrict__ b, float* __restrict__ ef)
{
  int idx = blockIdx.x * blockDim.x + threadIdx.x;
  if (idx >= EE * DD) return;
  int d = idx & 127, e = idx >> 7;
  float s = b[d];
#pragma unroll
  for (int j = 0; j < 4; ++j) s += ea[e * 4 + j] * w[j * DD + d];
  ef[idx] = fmaxf(s, 0.f);
}
__global__ void k_colmean(const float* __restrict__ ef, float* __restrict__ cm)
{
  int d = blockIdx.x;
  float s = 0.f;
  for (int e = threadIdx.x; e < EE; e += blockDim.x) s += ef[(size_t)e * DD + d];
  __shared__ float sh[256];
  sh[threadIdx.x] = s; __syncthreads();
  for (int o = 128; o > 0; o >>= 1) {
    if (threadIdx.x < o) sh[threadIdx.x] += sh[threadIdx.x + o];
    __syncthreads();
  }
  if (threadIdx.x == 0) cm[d] = sh[0] / (float)EE;
}
__global__ void k_fill_mean(const float* __restrict__ cm, float* __restrict__ ef)
{
  int idx = blockIdx.x * blockDim.x + threadIdx.x;
  if (idx >= NN * DD) return;
  ef[(size_t)EE * DD + idx] = cm[idx & 127];
}

__global__ void k_fill(unsigned* p, unsigned v, int n)
{
  int i = blockIdx.x * blockDim.x + threadIdx.x;
  if (i < n) p[i] = v;
}

// ---------------------------------------------------------------------------
// GAT edge kernels
// ---------------------------------------------------------------------------
__global__ void k_edge_logit(const float* __restrict__ xl, const float* __restrict__ xr,
                             const float* __restrict__ eh, const int* __restrict__ eix,
                             const float* __restrict__ att, float* __restrict__ logit,
                             unsigned* __restrict__ mx)
{
  int idx = blockIdx.x * blockDim.x + threadIdx.x;
  if (idx >= BW * ET * HH) return;
  int h = idx & 3; int be = idx >> 2; int e = be % ET; int bw = be / ET;
  int s = (e < EE) ? eix[e]      : e - EE;
  int d = (e < EE) ? eix[EE + e] : e - EE;
  const float* xls = xl + ((size_t)bw * NN + s) * DD + h * FHH;
  const float* xrd = xr + ((size_t)bw * NN + d) * DD + h * FHH;
  const float* ehr = eh + (size_t)e * DD + h * FHH;
  float acc = 0.f;
#pragma unroll 8
  for (int f = 0; f < FHH; ++f) {
    float sv = xls[f] + xrd[f] + ehr[f];
    sv = sv > 0.f ? sv : 0.2f * sv;             // leaky_relu(0.2)
    acc += sv * att[h * FHH + f];
  }
  logit[idx] = acc;
  unsigned ub  = __float_as_uint(acc);
  unsigned key = (ub & 0x80000000u) ? ~ub : (ub | 0x80000000u);  // order-preserving
  atomicMax(&mx[((size_t)bw * NN + d) * HH + h], key);
}
__global__ void k_decode_mx(unsigned* mx)
{
  int i = blockIdx.x * blockDim.x + threadIdx.x;
  if (i >= BW * NN * HH) return;
  unsigned key  = mx[i];
  unsigned bits = (key & 0x80000000u) ? (key & 0x7FFFFFFFu) : ~key;
  ((float*)mx)[i] = __uint_as_float(bits);
}
__global__ void k_edge_exp(float* __restrict__ logit, const float* __restrict__ mxf,
                           const int* __restrict__ eix, float* __restrict__ den)
{
  int idx = blockIdx.x * blockDim.x + threadIdx.x;
  if (idx >= BW * ET * HH) return;
  int h = idx & 3; int be = idx >> 2; int e = be % ET; int bw = be / ET;
  int d = (e < EE) ? eix[EE + e] : e - EE;
  float ex = __expf(logit[idx] - mxf[((size_t)bw * NN + d) * HH + h]);
  logit[idx] = ex;
  atomicAdd(&den[((size_t)bw * NN + d) * HH + h], ex);
}
__global__ void k_edge_scatter(const float* __restrict__ ex, const float* __restrict__ den,
                               const float* __restrict__ xl, const int* __restrict__ eix,
                               float* __restrict__ acc)
{
  long idx = blockIdx.x * (long)blockDim.x + threadIdx.x;
  if (idx >= (long)BW * ET * HH * FHH) return;
  int f = idx & 31; long r = idx >> 5;
  int h = r & 3; long be = r >> 2; int e = (int)(be % ET); int bw = (int)(be / ET);
  int s = (e < EE) ? eix[e]      : e - EE;
  int d = (e < EE) ? eix[EE + e] : e - EE;
  float alpha = ex[r] / den[((size_t)bw * NN + d) * HH + h];
  atomicAdd(&acc[((size_t)bw * NN + d) * DD + h * FHH + f],
            alpha * xl[((size_t)bw * NN + s) * DD + h * FHH + f]);
}

// GroupNorm over (B,N) per (w,d): single pass sum/sumsq
__global__ void k_gn_stats(const float* __restrict__ acc, const float* __restrict__ bias,
                           const float* __restrict__ ms, float* __restrict__ mu,
                           float* __restrict__ var)
{
  int wd = blockIdx.x; int w = wd >> 7; int d = wd & 127;
  float s = 0.f, q = 0.f;
  for (int i = threadIdx.x; i < BB * NN; i += blockDim.x) {
    int b = i / NN, n = i - b * NN;
    float v = acc[(((size_t)b * WW + w) * NN + n) * DD + d] + bias[d];
    s += v; q += v * v;
  }
  __shared__ float sh[256], sq[256];
  sh[threadIdx.x] = s; sq[threadIdx.x] = q; __syncthreads();
  for (int o = 128; o > 0; o >>= 1) {
    if (threadIdx.x < o) { sh[threadIdx.x] += sh[threadIdx.x + o]; sq[threadIdx.x] += sq[threadIdx.x + o]; }
    __syncthreads();
  }
  if (threadIdx.x == 0) {
    float m = sh[0] / (float)(BB * NN), m2 = sq[0] / (float)(BB * NN);
    float v = ms[d];
    mu[wd]  = m;
    var[wd] = m2 - 2.f * v * m * m + v * v * m * m;   // E[(hn - ms*mu)^2]
  }
}
__global__ void k_gn_apply(const float* __restrict__ acc, const float* __restrict__ bias,
                           const float* __restrict__ ms, const float* __restrict__ gw,
                           const float* __restrict__ gb, const float* __restrict__ mu,
                           const float* __restrict__ var, float* __restrict__ h)
{
  int idx = blockIdx.x * blockDim.x + threadIdx.x;
  if (idx >= M40 * DD) return;
  int d = idx & 127; int row = idx >> 7;
  int w = (row / NN) % WW; int wd = w * DD + d;
  float v = acc[idx] + bias[d];
  float o = v - ms[d] * mu[wd];
  float r = gw[d] * o * rsqrtf(var[wd] + 1e-5f) + gb[d];
  h[idx] += fmaxf(r, 0.f);
}

// (B,W,N,D) -> (B*N, W, D)
__global__ void k_transpose(const float* __restrict__ h, float* __restrict__ hs)
{
  int idx = blockIdx.x * blockDim.x + threadIdx.x;
  if (idx >= M40 * DD) return;
  int d = idx & 127; int row = idx >> 7;
  int n = row % NN; int w = (row / NN) % WW; int b = row / (NN * WW);
  hs[(((size_t)b * NN + n) * WW + w) * DD + d] = h[idx];
}

// GRU gating (one timestep)
__global__ void k_gru_gate(const float* __restrict__ gx, const float* __restrict__ gh,
                           float* __restrict__ hprev, float* __restrict__ ys, int t)
{
  int idx = blockIdx.x * blockDim.x + threadIdx.x;
  if (idx >= MGRU * DD) return;
  int m = idx >> 7, d = idx & 127;
  size_t gxr = ((size_t)m * WW + t) * 384;
  size_t ghr = (size_t)m * 384;
  float r = sigm(gx[gxr + d]       + gh[ghr + d]);
  float z = sigm(gx[gxr + DD + d]  + gh[ghr + DD + d]);
  float n = tanhf(gx[gxr + 2*DD + d] + r * gh[ghr + 2*DD + d]);
  float hp = hprev[idx];
  float hn = (1.f - z) * n + z * hp;
  hprev[idx] = hn;
  ys[((size_t)m * WW + t) * DD + d] = hn;
}

// MHA over W=10: thread per (m,h,qi): softmax row + ctx row + store aw
__global__ void k_attn(const float* __restrict__ qkv, float* __restrict__ aw,
                       float* __restrict__ ctx)
{
  int idx = blockIdx.x * blockDim.x + threadIdx.x;
  if (idx >= MGRU * HH * WW) return;
  int qi = idx % WW; int mh = idx / WW; int h = mh & 3; int m = mh >> 2;
  const float scale = 0.17677669529663687f;     // 1/sqrt(32)
  const float* qrow = qkv + ((size_t)m * WW + qi) * 384 + h * FHH;
  float lg[WW], mxv = -3.0e38f;
  for (int k = 0; k < WW; ++k) {
    const float* krow = qkv + ((size_t)m * WW + k) * 384 + DD + h * FHH;
    float s = 0.f;
#pragma unroll 8
    for (int f = 0; f < FHH; ++f) s += qrow[f] * krow[f];
    s *= scale; lg[k] = s; mxv = fmaxf(mxv, s);
  }
  float dn = 0.f;
  for (int k = 0; k < WW; ++k) { lg[k] = __expf(lg[k] - mxv); dn += lg[k]; }
  float inv = 1.f / dn;
  for (int k = 0; k < WW; ++k) {
    lg[k] *= inv;
    aw[(((size_t)m * HH + h) * WW + qi) * WW + k] = lg[k];
  }
  for (int f = 0; f < FHH; ++f) {
    float s = 0.f;
    for (int k = 0; k < WW; ++k)
      s += lg[k] * qkv[((size_t)m * WW + k) * 384 + 2*DD + h * FHH + f];
    ctx[((size_t)m * WW + qi) * DD + h * FHH + f] = s;
  }
}
__global__ void k_attn_mean(const float* __restrict__ aw, float* __restrict__ out)
{
  int idx = blockIdx.x * blockDim.x + threadIdx.x;
  if (idx >= MGRU * WW * WW) return;
  int qk = idx % (WW * WW); int m = idx / (WW * WW);
  float s = 0.f;
#pragma unroll
  for (int h = 0; h < HH; ++h) s += aw[(((size_t)m * HH + h) * WW * WW) + qk];
  out[idx] = s * 0.25f;
}
__global__ void k_gather_last(const float* __restrict__ ht, float* __restrict__ hf)
{
  int idx = blockIdx.x * blockDim.x + threadIdx.x;
  if (idx >= MGRU * DD) return;
  int d = idx & 127; int m = idx >> 7;
  hf[idx] = ht[((size_t)m * WW + (WW - 1)) * DD + d];
}

// Small head layer 2: act 0=none 1=relu 2=sigmoid 3=softmax(Nout)
__global__ void k_head2(const float* __restrict__ x, const float* __restrict__ W2,
                        const float* __restrict__ b2, float* __restrict__ out,
                        int M, int K, int Nout, int act)
{
  int m = blockIdx.x * blockDim.x + threadIdx.x;
  if (m >= M) return;
  float o[4];
  for (int j = 0; j < Nout; ++j) {
    float s = b2[j];
    for (int k = 0; k < K; ++k) s += x[(size_t)m * K + k] * W2[k * Nout + j];
    o[j] = s;
  }
  if (act == 1)      { for (int j = 0; j < Nout; ++j) o[j] = fmaxf(o[j], 0.f); }
  else if (act == 2) { for (int j = 0; j < Nout; ++j) o[j] = sigm(o[j]); }
  else if (act == 3) {
    float mx = o[0];
    for (int j = 1; j < Nout; ++j) mx = fmaxf(mx, o[j]);
    float dn = 0.f;
    for (int j = 0; j < Nout; ++j) { o[j] = __expf(o[j] - mx); dn += o[j]; }
    for (int j = 0; j < Nout; ++j) o[j] /= dn;
  }
  for (int j = 0; j < Nout; ++j) out[(size_t)m * Nout + j] = o[j];
}

// ---------------------------------------------------------------------------
// Host side
// ---------------------------------------------------------------------------
static inline int cdivl(long a, long b) { return (int)((a + b - 1) / b); }
#define L(kern, n, ...) kern<<<cdivl((long)(n), 256L), 256, 0, stream>>>(__VA_ARGS__)

extern "C" void kernel_launch(void* const* d_in, const int* in_sizes, int n_in,
                              void* d_out, int out_size, void* d_ws, size_t ws_size,
                              hipStream_t stream)
{
  (void)in_sizes; (void)n_in; (void)out_size; (void)ws_size;
  // Input mapping: JAX pytree flatten (dict keys sorted, tuples/lists in order):
  // edge_attr, edge_index, params{conf,demand,ee_b,ee_w,exc,fe_b,fe_be,fe_g,fe_w,
  //   gat[2]{We,Wl,Wr,att,bias,bl,br,gn_b,gn_ms,gn_w}, gru[2]{Whh,Wih,bhh,bih},
  //   mha{in_b,in_w,out_b,out_w}, order, prop, tn_b, tn_g},
  // structural_embeddings, x_temporal
  const float* edge_attr = (const float*)d_in[0];
  const int*   eidx      = (const int*)  d_in[1];
  const float *conf_W1=(const float*)d_in[2],  *conf_b1=(const float*)d_in[3],
              *conf_W2=(const float*)d_in[4],  *conf_b2=(const float*)d_in[5];
  const float *dem_W1 =(const float*)d_in[6],  *dem_b1 =(const float*)d_in[7],
              *dem_W2 =(const float*)d_in[8],  *dem_b2 =(const float*)d_in[9];
  const float *ee_b=(const float*)d_in[10], *ee_w=(const float*)d_in[11];
  const float *exc_W1=(const float*)d_in[12], *exc_b1=(const float*)d_in[13],
              *exc_W2=(const float*)d_in[14], *exc_b2=(const float*)d_in[15];
  const float *fe_b=(const float*)d_in[16], *fe_be=(const float*)d_in[17],
              *fe_g=(const float*)d_in[18], *fe_w =(const float*)d_in[19];
  struct GatP { const float *We,*Wl,*Wr,*att,*bias,*bl,*br,*gn_b,*gn_ms,*gn_w; } gat[2];
  for (int l = 0; l < 2; ++l) {
    int o = 20 + l * 10;
    gat[l] = { (const float*)d_in[o+0],(const float*)d_in[o+1],(const float*)d_in[o+2],
               (const float*)d_in[o+3],(const float*)d_in[o+4],(const float*)d_in[o+5],
               (const float*)d_in[o+6],(const float*)d_in[o+7],(const float*)d_in[o+8],
               (const float*)d_in[o+9] };
  }
  struct GruP { const float *Whh,*Wih,*bhh,*bih; } gru[2];
  for (int l = 0; l < 2; ++l) {
    int o = 40 + l * 4;
    gru[l] = { (const float*)d_in[o+0],(const float*)d_in[o+1],
               (const float*)d_in[o+2],(const float*)d_in[o+3] };
  }
  const float *in_b=(const float*)d_in[48], *in_w =(const float*)d_in[49],
              *out_b=(const float*)d_in[50],*out_w=(const float*)d_in[51];
  const float *ord_W1=(const float*)d_in[52], *ord_b1=(const float*)d_in[53],
              *ord_W2=(const float*)d_in[54], *ord_b2=(const float*)d_in[55];
  const float *prop_W1=(const float*)d_in[56], *prop_b1=(const float*)d_in[57],
              *prop_W2=(const float*)d_in[58], *prop_b2=(const float*)d_in[59];
  const float *tn_b=(const float*)d_in[60], *tn_g=(const float*)d_in[61];
  const float *se=(const float*)d_in[62], *xt=(const float*)d_in[63];
  float* out = (float*)d_out;

  // ---- workspace arena (~184 MB) ----
  char* base = (char*)d_ws;
  size_t cur = 0;
  auto alloc = [&](size_t bytes) -> char* {
    char* p = base + cur;
    cur = (cur + bytes + 255) & ~(size_t)255;
    return p;
  };
  float* Hbuf  = (float*)alloc((size_t)M40 * DD * 4);
  float* R1    = (float*)alloc((size_t)M40 * 384 * 4);  // xl|xr|acc, later gx/qkv/hattn
  float* XL = R1;
  float* XR = R1 + (size_t)M40 * DD;
  float* ACC = R1 + (size_t)2 * M40 * DD;
  float* GX = R1; float* QKV = R1; float* HATTN = R1;
  float* HSEQ  = (float*)alloc((size_t)M40 * DD * 4);
  float* HSEQ2 = (float*)alloc((size_t)M40 * DD * 4);
  float* EFULL = (float*)alloc((size_t)ET * DD * 4);
  float* EH    = (float*)alloc((size_t)ET * DD * 4);
  float* LOGIT = (float*)alloc((size_t)BW * ET * HH * 4);
  unsigned* MX = (unsigned*)alloc((size_t)BW * NN * HH * 4);
  float* DEN   = (float*)alloc((size_t)BW * NN * HH * 4);
  float* MUv   = (float*)alloc((size_t)WW * DD * 4);
  float* VARv  = (float*)alloc((size_t)WW * DD * 4);
  unsigned short* ABF  = (unsigned short*)alloc((size_t)40016 * DD * 2);
  unsigned short* BBF  = (unsigned short*)alloc((size_t)384 * DD * 2);
  float* HPREV = (float*)alloc((size_t)MGRU * DD * 4);
  float* GH    = (float*)alloc((size_t)MGRU * 384 * 4);
  unsigned short* HPBF = (unsigned short*)alloc((size_t)MGRU * DD * 2);
  float* AW    = (float*)alloc((size_t)MGRU * HH * WW * WW * 4);
  float* HF    = (float*)alloc((size_t)MGRU * DD * 4);
  unsigned short* HFBF = (unsigned short*)alloc((size_t)MGRU * DD * 2);
  float* HT2   = (float*)alloc((size_t)MGRU * DD * 4);
  float* CM    = (float*)alloc((size_t)DD * 4);

  auto gemm = [&](const unsigned short* A, const unsigned short* Bt, float* C,
                  const float* bias, int M, int N, int Kp, int relu) {
    int Mtl = (M + 15) / 16;
    if (N % 64 == 0) {
      int grps = Mtl * (N / 64);
      k_wmma_gemm<4><<<cdivl(grps, 4), 128, 0, stream>>>(A, Bt, C, bias, M, N, Kp, relu);
    } else {                                   // N == 32 head
      int grps = Mtl * (N / 32);
      k_wmma_gemm<2><<<cdivl(grps, 4), 128, 0, stream>>>(A, Bt, C, bias, M, N, Kp, relu);
    }
  };

  // ---- Phase 0: feature embedding ----
  L(k_build_x, M40 * 96, xt, se, ABF);
  L(k_conv_w_t, DD * 96, fe_w, BBF, 72, DD, 96);
  gemm(ABF, BBF, HSEQ, fe_b, M40, DD, 96, 0);
  k_ln<<<cdivl(M40, 8), 256, 0, stream>>>(HSEQ, nullptr, Hbuf, fe_g, fe_be, M40, 1);

  // ---- Phase 1: edge embedding + mean row ----
  L(k_edge_embed, EE * DD, edge_attr, ee_w, ee_b, EFULL);
  k_colmean<<<DD, 256, 0, stream>>>(EFULL, CM);
  L(k_fill_mean, NN * DD, CM, EFULL);

  // ---- Phase 2: two GAT layers ----
  for (int l = 0; l < 2; ++l) {
    const GatP& g = gat[l];
    L(k_conv_a, (long)M40 * DD, Hbuf, ABF, M40, M40, DD, DD);
    L(k_conv_w_t, DD * DD, g.Wl, BBF, DD, DD, DD);
    gemm(ABF, BBF, XL, g.bl, M40, DD, DD, 0);
    L(k_conv_w_t, DD * DD, g.Wr, BBF, DD, DD, DD);
    gemm(ABF, BBF, XR, g.br, M40, DD, DD, 0);
    L(k_conv_a, (long)ETP * DD, EFULL, ABF, ET, ETP, DD, DD);
    L(k_conv_w_t, DD * DD, g.We, BBF, DD, DD, DD);
    gemm(ABF, BBF, EH, nullptr, ET, DD, DD, 0);
    L(k_fill, BW * NN * HH, MX, 0u, BW * NN * HH);
    L(k_fill, BW * NN * HH, (unsigned*)DEN, 0u, BW * NN * HH);
    L(k_fill, M40 * DD, (unsigned*)ACC, 0u, M40 * DD);
    L(k_edge_logit, BW * ET * HH, XL, XR, EH, eidx, g.att, LOGIT, MX);
    L(k_decode_mx, BW * NN * HH, MX);
    L(k_edge_exp, BW * ET * HH, LOGIT, (const float*)MX, eidx, DEN);
    L(k_edge_scatter, (long)BW * ET * HH * FHH, LOGIT, DEN, XL, eidx, ACC);
    k_gn_stats<<<WW * DD, 256, 0, stream>>>(ACC, g.bias, g.gn_ms, MUv, VARv);
    L(k_gn_apply, M40 * DD, ACC, g.bias, g.gn_ms, g.gn_w, g.gn_b, MUv, VARv, Hbuf);
  }

  // ---- Phase 3: two GRU layers ----
  L(k_transpose, M40 * DD, Hbuf, HSEQ);
  for (int l = 0; l < 2; ++l) {
    const float* in = (l == 0) ? HSEQ : HSEQ2;
    float* outp     = (l == 0) ? HSEQ2 : HSEQ;
    L(k_conv_a, (long)M40 * DD, in, ABF, M40, M40, DD, DD);
    L(k_conv_w, 384 * DD, gru[l].Wih, BBF, 384, DD, DD);
    gemm(ABF, BBF, GX, gru[l].bih, M40, 384, DD, 0);
    L(k_fill, MGRU * DD, (unsigned*)HPREV, 0u, MGRU * DD);
    L(k_conv_w, 384 * DD, gru[l].Whh, BBF, 384, DD, DD);
    for (int t = 0; t < WW; ++t) {
      L(k_conv_a, (long)MGRU * DD, HPREV, HPBF, MGRU, MGRU, DD, DD);
      gemm(HPBF, BBF, GH, gru[l].bhh, MGRU, 384, DD, 0);
      L(k_gru_gate, MGRU * DD, GX, GH, HPREV, outp, t);
    }
  }

  // ---- Phase 4: temporal MHA + layernorm ----
  L(k_conv_a, (long)M40 * DD, HSEQ, ABF, M40, M40, DD, DD);
  L(k_conv_w, 384 * DD, in_w, BBF, 384, DD, DD);
  gemm(ABF, BBF, QKV, in_b, M40, 384, DD, 0);
  L(k_attn, MGRU * HH * WW, QKV, AW, HSEQ2);          // ctx -> HSEQ2
  L(k_conv_a, (long)M40 * DD, HSEQ2, ABF, M40, M40, DD, DD);
  L(k_conv_w, DD * DD, out_w, BBF, DD, DD, DD);
  gemm(ABF, BBF, HATTN, out_b, M40, DD, DD, 0);
  k_ln<<<cdivl(M40, 8), 256, 0, stream>>>(HATTN, HSEQ, HSEQ2, tn_g, tn_b, M40, 0);
  L(k_gather_last, MGRU * DD, HSEQ2, HF);
  L(k_attn_mean, MGRU * WW * WW, AW, out + 52000);

  // ---- Phase 5: heads ----
  L(k_conv_a, (long)MGRU * DD, HF, HFBF, MGRU, MGRU, DD, DD);
  // order: relu(relu(hf@W1+b1)@W2+b2) -> (4,1000,1)
  L(k_conv_w_t, DD * DD, ord_W1, BBF, DD, DD, DD);
  gemm(HFBF, BBF, HT2, ord_b1, MGRU, DD, DD, 1);
  L(k_head2, MGRU, HT2, ord_W2, ord_b2, out + 0, MGRU, DD, 1, 1);
  // demand -> (4,1000,4)
  L(k_conv_w_t, DD * DD, dem_W1, BBF, DD, DD, DD);
  gemm(HFBF, BBF, HT2, dem_b1, MGRU, DD, DD, 1);
  L(k_head2, MGRU, HT2, dem_W2, dem_b2, out + 4000, MGRU, DD, 4, 0);
  // exc: softmax3 -> (4,1000,3)
  L(k_conv_w_t, 64 * DD, exc_W1, BBF, DD, 64, DD);
  gemm(HFBF, BBF, HT2, exc_b1, MGRU, 64, DD, 1);
  L(k_head2, MGRU, HT2, exc_W2, exc_b2, out + 20000, MGRU, 64, 3, 3);
  // prop: sigmoid -> (4,1000,4)
  L(k_conv_w_t, DD * DD, prop_W1, BBF, DD, DD, DD);
  gemm(HFBF, BBF, HT2, prop_b1, MGRU, DD, DD, 1);
  L(k_head2, MGRU, HT2, prop_W2, prop_b2, out + 32000, MGRU, DD, 4, 2);
  // conf: sigmoid -> (4,1000,1)
  L(k_conv_w_t, 32 * DD, conf_W1, BBF, DD, 32, DD);
  gemm(HFBF, BBF, HT2, conf_b1, MGRU, 32, DD, 1);
  L(k_head2, MGRU, HT2, conf_W2, conf_b2, out + 48000, MGRU, 32, 1, 2);
}
